// BahdanauSeq2Seq_88364657147913
// MI455X (gfx1250) — matrix operand order repaired and checked by reading, not compiled, not run on hardware
//
#include <hip/hip_runtime.h>
#include <hip/hip_bf16.h>

// ---- problem constants (from reference) ----
#define S_LEN 96
#define T_LEN 32
#define BATCH 32
#define DIN   512
#define HID   512
#define VOCAB 32000

typedef __attribute__((ext_vector_type(16))) __bf16 v16bf;
typedef __attribute__((ext_vector_type(8)))  float  v8f;

union Frag32B { v16bf v; uint4 q[2]; };

__device__ __forceinline__ float sigm(float x) { return 1.0f / (1.0f + __expf(-x)); }

// ---------------- conversion ----------------
__global__ void cvt_f32_bf16(const float* __restrict__ in,
                             __hip_bfloat16* __restrict__ out, int n) {
    int i = blockIdx.x * blockDim.x + threadIdx.x;
    if (i < n) out[i] = __float2bfloat16(in[i]);
}

// ---------------- fragment loaders (contiguous K order k = 16*half + s; A and B match) ----------------
// A fragment (16x32 bf16, row m = lane&15):
//   elems 0..7  : K = kb + half*16 + 0..7
//   elems 8..15 : K = kb + half*16 + 8 + 0..7
__device__ __forceinline__ v16bf load_a_frag(const __hip_bfloat16* __restrict__ A,
                                             int lda, int m, int kb, int half) {
    Frag32B a;
    a.q[0] = *reinterpret_cast<const uint4*>(A + (size_t)m * lda + kb + half * 16);
    a.q[1] = *reinterpret_cast<const uint4*>(A + (size_t)m * lda + kb + half * 16 + 8);
    return a.v;
}
// B fragment (32x16, col n = lane&15): 16 contiguous K at kb + half*16, read from
// row-major W (NxK) so B[k][n] = W[n][k].
__device__ __forceinline__ v16bf load_b_frag(const __hip_bfloat16* __restrict__ W,
                                             int ldw, int n, int kb, int half) {
    Frag32B b;
    const uint4* wp = reinterpret_cast<const uint4*>(W + (size_t)n * ldw + kb + half * 16);
    b.q[0] = wp[0];
    b.q[1] = wp[1];
    return b.v;
}
// B fragment from an LDS slab Bs[64][128] bf16 (stored as uint4 quads, 16 quads/row):
__device__ __forceinline__ v16bf lds_b_frag(const uint4* Bs, int nLocal, int kc, int half) {
    Frag32B b;
    const int q0 = nLocal * 16 + kc * 4 + half * 2;   // 16 contiguous K = 2 quads
    b.q[0] = Bs[q0];
    b.q[1] = Bs[q0 + 1];
    return b.v;
}

// ---------------- generic dual-A bf16 WMMA GEMM (1 tile / wave) ----------------
// D(MxN) = A1(MxK1) @ W1(NxK1)^T [+ A2(MxK2) @ W2(NxK2)^T] + bias1 + bias2
__global__ __launch_bounds__(256) void gemm_bf16_wmma(
        const __hip_bfloat16* __restrict__ A1, int K1, const __hip_bfloat16* __restrict__ W1,
        const __hip_bfloat16* __restrict__ A2, int K2, const __hip_bfloat16* __restrict__ W2,
        const float* __restrict__ bias1, const float* __restrict__ bias2,
        float* __restrict__ D, __hip_bfloat16* __restrict__ Dbf, int M, int N) {
    const int wid  = (int)((blockIdx.x * blockDim.x + threadIdx.x) >> 5);
    const int lane = (int)(threadIdx.x & 31);
    const int mt = M >> 4;
    const int nt = N >> 4;
    if (wid >= mt * nt) return;               // uniform per wave, EXEC stays all-ones
    const int tM = wid % mt, tN = wid / mt;
    const int half = lane >> 4, r = lane & 15;
    const int m = tM * 16 + r;                // A row handled by this lane
    const int n = tN * 16 + r;                // W row / output column
    v8f acc = {};
    for (int kb = 0; kb < K1; kb += 32) {
        v16bf a = load_a_frag(A1, K1, m, kb, half);
        v16bf b = load_b_frag(W1, K1, n, kb, half);
        acc = __builtin_amdgcn_wmma_f32_16x16x32_bf16(false, a, false, b,
                                                      (short)0, acc, false, false);
    }
    if (A2)
        for (int kb = 0; kb < K2; kb += 32) {
            v16bf a = load_a_frag(A2, K2, m, kb, half);
            v16bf b = load_b_frag(W2, K2, n, kb, half);
            acc = __builtin_amdgcn_wmma_f32_16x16x32_bf16(false, a, false, b,
                                                          (short)0, acc, false, false);
        }
    float bn = 0.0f;
    if (bias1) bn += bias1[n];
    if (bias2) bn += bias2[n];
#pragma unroll
    for (int j = 0; j < 8; ++j) {             // D: VGPR j holds row tM*16 + half*8 + j
        const int mo = tM * 16 + half * 8 + j;
        const float v = acc[j] + bn;
        D[(size_t)mo * N + n] = v;
        if (Dbf) Dbf[(size_t)mo * N + n] = __float2bfloat16(v);
    }
}

// ------- LDS-staged register-blocked bf16 WMMA GEMM: 2M x 4N tiles per wave -------
// D(MxN) = A(MxK) @ W(NxK)^T + bias.  Requires M%32==0, N%64==0, K%128==0.
// Block = 8 waves sharing one 64-column N slice; W slab (64x128, 16KB) staged in LDS
// once per 128-K step and consumed by all 8 waves (8x reuse). A streams from L2.
__global__ __launch_bounds__(256) void gemm_bf16_wmma_blk(
        const __hip_bfloat16* __restrict__ A, int K, const __hip_bfloat16* __restrict__ W,
        const float* __restrict__ bias, float* __restrict__ D, int M, int N) {
    __shared__ uint4 Bs[1024];                // 64 rows x 128 K bf16 = 16 KB
    const int tid  = (int)threadIdx.x;
    const int lane = tid & 31;
    const int wv   = tid >> 5;                // wave in block: 0..7
    const int mj   = M >> 5;                  // #M jobs (32 rows each)
    const int tM   = (int)blockIdx.x * 8 + wv;
    const int nBase = (int)blockIdx.y * 64;
    const bool active = (tM < mj);            // wave-uniform; idle waves still barrier
    const int half = lane >> 4, r = lane & 15;
    const int m0 = tM * 32 + r;
    v8f acc00 = {}, acc01 = {}, acc02 = {}, acc03 = {};
    v8f acc10 = {}, acc11 = {}, acc12 = {}, acc13 = {};
    for (int kslab = 0; kslab < K; kslab += 128) {
        __syncthreads();                      // previous slab fully consumed
#pragma unroll
        for (int q = 0; q < 4; ++q) {         // cooperative stage: 1024 quads / 256 thr
            const int chunk = tid + q * 256;
            const int row  = chunk >> 4;      // 16 quads per 128-elem row
            const int colq = chunk & 15;
            const __hip_bfloat16* gp = W + (size_t)(nBase + row) * K + kslab + colq * 8;
            if (kslab + 128 < K)              // keep W stream ahead in GL2
                __builtin_prefetch(gp + 128, 0, 1);
            Bs[chunk] = *reinterpret_cast<const uint4*>(gp);
        }
        __syncthreads();
        if (active) {
#pragma unroll
            for (int kc = 0; kc < 4; ++kc) {
                const int kb = kslab + kc * 32;
                const v16bf a0 = load_a_frag(A, K, m0, kb, half);
                const v16bf a1 = load_a_frag(A, K, m0 + 16, kb, half);
                const v16bf b0 = lds_b_frag(Bs, r, kc, half);
                const v16bf b1 = lds_b_frag(Bs, 16 + r, kc, half);
                const v16bf b2 = lds_b_frag(Bs, 32 + r, kc, half);
                const v16bf b3 = lds_b_frag(Bs, 48 + r, kc, half);
                acc00 = __builtin_amdgcn_wmma_f32_16x16x32_bf16(false, a0, false, b0, (short)0, acc00, false, false);
                acc01 = __builtin_amdgcn_wmma_f32_16x16x32_bf16(false, a0, false, b1, (short)0, acc01, false, false);
                acc02 = __builtin_amdgcn_wmma_f32_16x16x32_bf16(false, a0, false, b2, (short)0, acc02, false, false);
                acc03 = __builtin_amdgcn_wmma_f32_16x16x32_bf16(false, a0, false, b3, (short)0, acc03, false, false);
                acc10 = __builtin_amdgcn_wmma_f32_16x16x32_bf16(false, a1, false, b0, (short)0, acc10, false, false);
                acc11 = __builtin_amdgcn_wmma_f32_16x16x32_bf16(false, a1, false, b1, (short)0, acc11, false, false);
                acc12 = __builtin_amdgcn_wmma_f32_16x16x32_bf16(false, a1, false, b2, (short)0, acc12, false, false);
                acc13 = __builtin_amdgcn_wmma_f32_16x16x32_bf16(false, a1, false, b3, (short)0, acc13, false, false);
            }
        }
    }
    if (active) {
        v8f* accs[2][4] = {{&acc00, &acc01, &acc02, &acc03},
                           {&acc10, &acc11, &acc12, &acc13}};
#pragma unroll
        for (int u = 0; u < 4; ++u) {
            const int n = nBase + u * 16 + r;
            const float bn = bias ? bias[n] : 0.0f;
#pragma unroll
            for (int t = 0; t < 2; ++t) {
                const v8f& a = *accs[t][u];
#pragma unroll
                for (int j = 0; j < 8; ++j) {
                    const int mo = tM * 32 + t * 16 + half * 8 + j;
                    D[(size_t)mo * N + n] = a[j] + bn;
                }
            }
        }
    }
}

// ---------------- LSTM cell (gate order i,f,g,o) ----------------
__global__ void lstm_cell(const float* __restrict__ gates,   // B x 4H
                          float* __restrict__ c,             // B x H (in/out, fp32)
                          __hip_bfloat16* __restrict__ h_bf, // B x H (out, bf16)
                          __hip_bfloat16* __restrict__ h2,   // optional extra copy
                          int h2_stride) {
    int id = blockIdx.x * blockDim.x + threadIdx.x;
    if (id >= BATCH * HID) return;
    const int b = id / HID, j = id % HID;
    const float* g = gates + (size_t)b * 4 * HID;
    const float ig = sigm(g[j]);
    const float fg = sigm(g[HID + j]);
    const float gg = tanhf(g[2 * HID + j]);
    const float og = sigm(g[3 * HID + j]);
    const float cn = fg * c[id] + ig * gg;
    c[id] = cn;
    const __hip_bfloat16 hb = __float2bfloat16(og * tanhf(cn));
    h_bf[id] = hb;
    if (h2) h2[(size_t)b * h2_stride + j] = hb;
}

// ---------------- attention ----------------
// scores[s,b] = sum_h tanh(enc_proj[s,b,h] + dec_proj[b,h]) * v[h]; one wave per (s,b)
__global__ void attn_scores(const float* __restrict__ enc_proj,
                            const float* __restrict__ dec_proj,
                            const float* __restrict__ v_att,
                            float* __restrict__ scores) {
    const int wave = (int)((blockIdx.x * blockDim.x + threadIdx.x) >> 5);
    const int lane = (int)(threadIdx.x & 31);
    if (wave >= S_LEN * BATCH) return;
    const int b = wave % BATCH;
    const float* ep = enc_proj + (size_t)wave * HID;
    const float* dp = dec_proj + (size_t)b * HID;
    float acc = 0.0f;
    for (int h = lane; h < HID; h += 32)
        acc += tanhf(ep[h] + dp[h]) * v_att[h];
    for (int off = 16; off; off >>= 1) acc += __shfl_xor(acc, off, 32);
    if (lane == 0) scores[wave] = acc;
}

// softmax over s for each b (one thread per batch element, one wave total)
__global__ void softmax_s(const float* __restrict__ scores, float* __restrict__ attn) {
    const int b = threadIdx.x;
    if (b >= BATCH) return;
    float mx = -1e30f;
    for (int s = 0; s < S_LEN; ++s) mx = fmaxf(mx, scores[s * BATCH + b]);
    float sum = 0.0f;
    for (int s = 0; s < S_LEN; ++s) {
        const float e = __expf(scores[s * BATCH + b] - mx);
        attn[s * BATCH + b] = e;
        sum += e;
    }
    const float inv = 1.0f / sum;
    for (int s = 0; s < S_LEN; ++s) attn[s * BATCH + b] *= inv;
}

// context[b,d] = sum_s attn[s,b] * enc_out[s,b,d]; write bf16 into xcat[:, 512:1536]
__global__ void attn_context(const float* __restrict__ attn,
                             const __hip_bfloat16* __restrict__ enc_out,
                             __hip_bfloat16* __restrict__ xcat) {
    int id = blockIdx.x * blockDim.x + threadIdx.x;
    if (id >= BATCH * 2 * HID) return;
    const int b = id / (2 * HID), d = id % (2 * HID);
    float acc = 0.0f;
    for (int s = 0; s < S_LEN; ++s)
        acc += attn[s * BATCH + b] *
               __bfloat162float(enc_out[((size_t)(s * BATCH + b)) * (2 * HID) + d]);
    xcat[(size_t)b * (DIN + 2 * HID) + DIN + d] = __float2bfloat16(acc);
}

// xcat[:, 0:512] = bf16(tgt[t])
__global__ void xcat_tgt(const float* __restrict__ tgt_t, __hip_bfloat16* __restrict__ xcat) {
    int id = blockIdx.x * blockDim.x + threadIdx.x;
    if (id >= BATCH * DIN) return;
    const int b = id / DIN, j = id % DIN;
    xcat[(size_t)b * (DIN + 2 * HID) + j] = __float2bfloat16(tgt_t[(size_t)b * DIN + j]);
}

// henc[b, 0:H] = y1[S-1, b, 0:H] (fwd final), henc[b, H:2H] = y1[0, b, H:2H] (bwd final)
__global__ void gather_henc(const __hip_bfloat16* __restrict__ y1,
                            __hip_bfloat16* __restrict__ henc) {
    int id = blockIdx.x * blockDim.x + threadIdx.x;
    if (id >= BATCH * 2 * HID) return;
    const int b = id / (2 * HID), j = id % (2 * HID);
    henc[id] = (j < HID) ? y1[((size_t)(S_LEN - 1) * BATCH + b) * (2 * HID) + j]
                         : y1[((size_t)b) * (2 * HID) + j];
}

// =====================================================================
extern "C" void kernel_launch(void* const* d_in, const int* in_sizes, int n_in,
                              void* d_out, int out_size, void* d_ws, size_t ws_size,
                              hipStream_t stream) {
    (void)in_sizes; (void)n_in; (void)out_size; (void)ws_size;
    // ---- inputs (setup_inputs dict order, leaves flattened) ----
    const float* src = (const float*)d_in[0];   // (S,B,DIN)
    const float* tgt = (const float*)d_in[1];   // (T,B,DIN)
    const float* eW[2][2][4];                   // [layer][dir]{Wih,Whh,bih,bhh}
    int idx = 2;
    for (int l = 0; l < 2; ++l)
        for (int d = 0; d < 2; ++d)
            for (int p = 0; p < 4; ++p) eW[l][d][p] = (const float*)d_in[idx++];
    const float* dW[2][4];                      // [layer]{Wih,Whh,bih,bhh}
    for (int l = 0; l < 2; ++l)
        for (int p = 0; p < 4; ++p) dW[l][p] = (const float*)d_in[idx++];
    const float* W_enc = (const float*)d_in[idx++];  // (H, 2H)
    const float* W_dec = (const float*)d_in[idx++];  // (H, H)
    const float* v_att = (const float*)d_in[idx++];  // (1, H)
    const float* W_out = (const float*)d_in[idx++];  // (V, H)
    const float* b_out = (const float*)d_in[idx++];  // (V,)
    const float* W_hid = (const float*)d_in[idx++];  // (H, 2H)
    const float* b_hid = (const float*)d_in[idx++];  // (H,)
    float* out = (float*)d_out;                      // (T-1, B, V)

    // ---- workspace carve-up ----
    char* base = (char*)d_ws;
    size_t off = 0;
    auto alloc = [&](size_t bytes) -> void* {
        off = (off + 255) & ~(size_t)255;
        void* p = base + off;
        off += bytes;
        return p;
    };
    typedef __hip_bfloat16 bf;
    const size_t G4H = (size_t)BATCH * 4 * HID;        // gates elems
    bf* src_bf = (bf*)alloc((size_t)S_LEN * BATCH * DIN * 2);
    bf* eWih_bf[2][2], *eWhh_bf[2][2];
    for (int l = 0; l < 2; ++l)
        for (int d = 0; d < 2; ++d) {
            const size_t kin = (l == 0) ? DIN : 2 * HID;
            eWih_bf[l][d] = (bf*)alloc((size_t)4 * HID * kin * 2);
            eWhh_bf[l][d] = (bf*)alloc((size_t)4 * HID * HID * 2);
        }
    bf* dWih0_bf = (bf*)alloc((size_t)4 * HID * (DIN + 2 * HID) * 2);
    bf* dWhh0_bf = (bf*)alloc((size_t)4 * HID * HID * 2);
    bf* dWih1_bf = (bf*)alloc((size_t)4 * HID * HID * 2);
    bf* dWhh1_bf = (bf*)alloc((size_t)4 * HID * HID * 2);
    bf* Wenc_bf  = (bf*)alloc((size_t)HID * 2 * HID * 2);
    bf* Wdec_bf  = (bf*)alloc((size_t)HID * HID * 2);
    bf* Wout_bf  = (bf*)alloc((size_t)VOCAB * HID * 2);
    bf* Whid_bf  = (bf*)alloc((size_t)HID * 2 * HID * 2);
    bf* y0 = (bf*)alloc((size_t)S_LEN * BATCH * 2 * HID * 2);
    bf* y1 = (bf*)alloc((size_t)S_LEN * BATCH * 2 * HID * 2);   // enc_out (bf16)
    float* enc_proj = (float*)alloc((size_t)S_LEN * BATCH * HID * 4);
    float* gates_f = (float*)alloc(G4H * 4);
    float* gates_b = (float*)alloc(G4H * 4);
    float* gates_d = (float*)alloc(G4H * 4);
    bf* h_f = (bf*)alloc((size_t)BATCH * HID * 2);
    bf* h_b = (bf*)alloc((size_t)BATCH * HID * 2);
    bf* h1  = (bf*)alloc((size_t)BATCH * HID * 2);
    bf* h2  = (bf*)alloc((size_t)BATCH * HID * 2);
    float* c_f = (float*)alloc((size_t)BATCH * HID * 4);
    float* c_b = (float*)alloc((size_t)BATCH * HID * 4);
    float* c1  = (float*)alloc((size_t)BATCH * HID * 4);
    float* c2  = (float*)alloc((size_t)BATCH * HID * 4);
    bf* henc = (bf*)alloc((size_t)BATCH * 2 * HID * 2);
    float* dech0 = (float*)alloc((size_t)BATCH * HID * 4);
    float* dproj = (float*)alloc((size_t)BATCH * HID * 4);
    float* scoresb = (float*)alloc((size_t)S_LEN * BATCH * 4);
    float* attnb   = (float*)alloc((size_t)S_LEN * BATCH * 4);
    bf* xcat = (bf*)alloc((size_t)BATCH * (DIN + 2 * HID) * 2);
    bf* hist = (bf*)alloc((size_t)(T_LEN - 1) * BATCH * HID * 2); // h2 history

    // ---- helpers ----
    auto CVT = [&](const float* fin, bf* bout, size_t n) {
        cvt_f32_bf16<<<(unsigned)((n + 255) / 256), 256, 0, stream>>>(fin, bout, (int)n);
    };
    auto GEMM = [&](const bf* A1, int K1, const bf* W1,
                    const bf* A2, int K2, const bf* W2,
                    const float* b1, const float* b2,
                    float* Dp, bf* Dbf, int M, int N) {
        const int tiles = (M / 16) * (N / 16);
        gemm_bf16_wmma<<<(tiles + 7) / 8, 256, 0, stream>>>(A1, K1, W1, A2, K2, W2,
                                                            b1, b2, Dp, Dbf, M, N);
    };
    auto GEMM_BIG = [&](const bf* A, int K, const bf* W, const float* b1,
                        float* Dp, int M, int N) {   // M%32==0, N%64==0, K%128==0
        const int mj = M / 32;
        dim3 grid((mj + 7) / 8, N / 64);
        gemm_bf16_wmma_blk<<<grid, 256, 0, stream>>>(A, K, W, b1, Dp, M, N);
    };
    auto CELL = [&](const float* g, float* c, bf* hb, bf* hb2, int stride2) {
        lstm_cell<<<(BATCH * HID + 255) / 256, 256, 0, stream>>>(g, c, hb, hb2, stride2);
    };

    // ---- weight / input conversions (bf16) ----
    CVT(src, src_bf, (size_t)S_LEN * BATCH * DIN);
    for (int l = 0; l < 2; ++l)
        for (int d = 0; d < 2; ++d) {
            const size_t kin = (l == 0) ? DIN : 2 * HID;
            CVT(eW[l][d][0], eWih_bf[l][d], (size_t)4 * HID * kin);
            CVT(eW[l][d][1], eWhh_bf[l][d], (size_t)4 * HID * HID);
        }
    CVT(dW[0][0], dWih0_bf, (size_t)4 * HID * (DIN + 2 * HID));
    CVT(dW[0][1], dWhh0_bf, (size_t)4 * HID * HID);
    CVT(dW[1][0], dWih1_bf, (size_t)4 * HID * HID);
    CVT(dW[1][1], dWhh1_bf, (size_t)4 * HID * HID);
    CVT(W_enc, Wenc_bf, (size_t)HID * 2 * HID);
    CVT(W_dec, Wdec_bf, (size_t)HID * HID);
    CVT(W_out, Wout_bf, (size_t)VOCAB * HID);
    CVT(W_hid, Whid_bf, (size_t)HID * 2 * HID);

    // ---- encoder: 2 layers, bidirectional ----
    for (int l = 0; l < 2; ++l) {
        const bf* xin = (l == 0) ? src_bf : y0;
        const int Kin = (l == 0) ? DIN : 2 * HID;
        bf* yout = (l == 0) ? y0 : y1;
        hipMemsetAsync(h_f, 0, (size_t)BATCH * HID * 2, stream);
        hipMemsetAsync(h_b, 0, (size_t)BATCH * HID * 2, stream);
        hipMemsetAsync(c_f, 0, (size_t)BATCH * HID * 4, stream);
        hipMemsetAsync(c_b, 0, (size_t)BATCH * HID * 4, stream);
        for (int t = 0; t < S_LEN; ++t) {
            const int tf = t, tb = S_LEN - 1 - t;
            GEMM(xin + (size_t)tf * BATCH * Kin, Kin, eWih_bf[l][0],
                 h_f, HID, eWhh_bf[l][0], eW[l][0][2], eW[l][0][3],
                 gates_f, nullptr, BATCH, 4 * HID);
            GEMM(xin + (size_t)tb * BATCH * Kin, Kin, eWih_bf[l][1],
                 h_b, HID, eWhh_bf[l][1], eW[l][1][2], eW[l][1][3],
                 gates_b, nullptr, BATCH, 4 * HID);
            CELL(gates_f, c_f, h_f, yout + (size_t)tf * BATCH * 2 * HID, 2 * HID);
            CELL(gates_b, c_b, h_b, yout + (size_t)tb * BATCH * 2 * HID + HID, 2 * HID);
        }
    }

    // enc_proj = enc_out @ W_enc^T : (S*B, H)   (M=3072, N=512, K=1024)
    GEMM_BIG(y1, 2 * HID, Wenc_bf, nullptr, enc_proj, S_LEN * BATCH, HID);

    // dec_h0 = concat(hf, hb) @ W_hid^T + b_hid
    gather_henc<<<(BATCH * 2 * HID + 255) / 256, 256, 0, stream>>>(y1, henc);
    GEMM(henc, 2 * HID, Whid_bf, nullptr, 0, nullptr, b_hid, nullptr,
         dech0, h1, BATCH, HID);
    hipMemcpyAsync(h2, h1, (size_t)BATCH * HID * 2, hipMemcpyDeviceToDevice, stream);
    hipMemsetAsync(c1, 0, (size_t)BATCH * HID * 4, stream);
    hipMemsetAsync(c2, 0, (size_t)BATCH * HID * 4, stream);

    // ---- decoder: T-1 steps; logits deferred to one big GEMM ----
    for (int t = 0; t < T_LEN - 1; ++t) {
        GEMM(h2, HID, Wdec_bf, nullptr, 0, nullptr, nullptr, nullptr,
             dproj, nullptr, BATCH, HID);
        attn_scores<<<(S_LEN * BATCH) / 4, 128, 0, stream>>>(enc_proj, dproj, v_att, scoresb);
        softmax_s<<<1, 32, 0, stream>>>(scoresb, attnb);
        attn_context<<<(BATCH * 2 * HID + 255) / 256, 256, 0, stream>>>(attnb, y1, xcat);
        xcat_tgt<<<(BATCH * DIN + 255) / 256, 256, 0, stream>>>(
            tgt + (size_t)t * BATCH * DIN, xcat);
        GEMM(xcat, DIN + 2 * HID, dWih0_bf, h1, HID, dWhh0_bf,
             dW[0][2], dW[0][3], gates_d, nullptr, BATCH, 4 * HID);
        CELL(gates_d, c1, h1, nullptr, 0);
        GEMM(h1, HID, dWih1_bf, h2, HID, dWhh1_bf,
             dW[1][2], dW[1][3], gates_d, nullptr, BATCH, 4 * HID);
        CELL(gates_d, c2, h2, hist + (size_t)t * BATCH * HID, HID);
    }

    // logits for all steps at once: (992 x 32000) = hist(992x512) @ W_out^T + b_out
    // (M=992=31*32, N=32000=500*64, K=512) -> LDS-staged register-blocked kernel
    GEMM_BIG(hist, HID, Wout_bf, b_out, out, (T_LEN - 1) * BATCH, VOCAB);
}
